// MaskPostProcessor_9045201125729
// MI455X (gfx1250) — matrix operand order repaired and checked by reading, not compiled, not run on hardware
//
#include <hip/hip_runtime.h>
#include <hip/hip_bf16.h>

typedef __attribute__((ext_vector_type(16))) _Float16 v16h;
typedef __attribute__((ext_vector_type(8)))  _Float16 v8h;
typedef __attribute__((ext_vector_type(8)))  float    v8f;

#define NMS_THRESH 0.5f
#define IOU_EPS    1e-4f

// ---------------------------------------------------------------------------
// 1) Stable descending rank (replaces argsort(-scores)): rank[i] = #(j) with
//    scores[j] > scores[i], ties broken by lower index. order[rank[i]] = i.
// ---------------------------------------------------------------------------
__global__ void rank_order_kernel(const float* __restrict__ scores,
                                  int* __restrict__ rank,
                                  int* __restrict__ order, int N) {
    int i = blockIdx.x * blockDim.x + threadIdx.x;
    if (i >= N) return;
    float si = scores[i];
    int r = 0;
    for (int j = 0; j < N; ++j) {
        float sj = scores[j];
        r += (sj > si) || (sj == si && j < i);
    }
    rank[i] = r;
    order[r] = i;
}

// ---------------------------------------------------------------------------
// 2) Gather selected channel, sigmoid, store f16 row-major (sorted order),
//    and reduce per-row area (sum of the f16-rounded probabilities so the
//    IoU numerator/denominator are consistent).
// ---------------------------------------------------------------------------
__global__ __launch_bounds__(256)
void gather_sigmoid_kernel(const float* __restrict__ x,
                           const int* __restrict__ labels,
                           const int* __restrict__ order,
                           _Float16* __restrict__ S,
                           float* __restrict__ area,
                           int C, int K) {
    __shared__ float red[256];
    int p = blockIdx.x;               // sorted position
    int n = order[p];
    int c = labels[n];
    const float* src = x + ((size_t)n * C + c) * K;
    _Float16* dst = S + (size_t)p * K;
    float sum = 0.0f;
    for (int k = threadIdx.x; k < K; k += 256) {
        float v = 1.0f / (1.0f + __expf(-src[k]));
        _Float16 h = (_Float16)v;
        dst[k] = h;
        sum += (float)h;
    }
    red[threadIdx.x] = sum;
    __syncthreads();
    for (int s = 128; s > 0; s >>= 1) {
        if (threadIdx.x < s) red[threadIdx.x] += red[threadIdx.x + s];
        __syncthreads();
    }
    if (threadIdx.x == 0) area[p] = red[0];
}

// ---------------------------------------------------------------------------
// 3) inter = S * S^T via v_wmma_f32_16x16x32_f16, one wave per 16x16 tile.
//    Fragments built per CDNA5 wave32 VGPR layouts (ISA 7.12.2):
//      A (16x32 f16): lane L holds M=L%16; halves e<8 -> K=(L/16)*8+e,
//                     e>=8 -> K=16+(L/16)*8+(e-8)  => two contiguous 8-half runs.
//      B (32x16 f16): lane L holds N=L%16; same K mapping as A
//                     => two contiguous 8-half runs.
//    Output normalized to iou[row][col] = inter / (area[row] + eps).
// ---------------------------------------------------------------------------
__global__ __launch_bounds__(32)
void inter_iou_wmma_kernel(const _Float16* __restrict__ S,
                           const float* __restrict__ area,
                           float* __restrict__ iou,
                           int N, int K) {
    const int lane  = threadIdx.x;      // 0..31
    const int hlf   = lane >> 4;        // 0 or 1
    const int l16   = lane & 15;
    const int i0    = blockIdx.y * 16;  // kept-row tile
    const int j0    = blockIdx.x * 16;  // candidate-col tile

    const _Float16* pa = S + (size_t)(i0 + l16) * K;
    const _Float16* pb = S + (size_t)(j0 + l16) * K;
    const int aOff0 = hlf * 8;
    const int aOff1 = 16 + hlf * 8;
    const int bOff  = hlf * 16;

    v8f c = {};
    for (int k0 = 0; k0 < K; k0 += 32) {
        union { v16h v; v8h h[2]; } a;
        a.h[0] = *(const v8h*)(pa + k0 + aOff0);
        a.h[1] = *(const v8h*)(pa + k0 + aOff1);
        union { v16h v; v8h h[2]; } b;
        b.h[0] = *(const v8h*)(pb + k0 + aOff0);
        b.h[1] = *(const v8h*)(pb + k0 + aOff1);
        if (k0 + 32 < K) {
            __builtin_prefetch(pb + k0 + 32 + bOff, 0, 1);
        }
        c = __builtin_amdgcn_wmma_f32_16x16x32_f16(
                /*neg_a=*/false, a.v, /*neg_b=*/false, b.v,
                /*c_mod=*/(short)0, c, /*reuse_a=*/false, /*reuse_b=*/false);
    }

    // C/D layout: VGPR r, lanes 0-15 -> M=r; lanes 16-31 -> M=8+r; N=lane%16.
    #pragma unroll
    for (int r = 0; r < 8; ++r) {
        int row = i0 + r + 8 * hlf;
        int col = j0 + l16;
        iou[(size_t)row * N + col] = c[r] / (area[row] + IOU_EPS);
    }
}

// ---------------------------------------------------------------------------
// 4) Greedy sequential NMS on one wave32; keep flags in LDS, map back to
//    original index order at the end.
// ---------------------------------------------------------------------------
__global__ __launch_bounds__(32)
void greedy_nms_kernel(const float* __restrict__ iou,
                       const int* __restrict__ order,
                       int* __restrict__ keepO, int N) {
    __shared__ int keep[1024];
    int lane = threadIdx.x;
    for (int t = lane; t < N; t += 32) keep[t] = 1;
    __syncthreads();
    for (int i = 1; i < N; ++i) {
        int sup = 0;
        for (int j = lane; j < i; j += 32)
            sup |= (keep[j] && (iou[(size_t)j * N + i] >= NMS_THRESH));
        int anysup = __any(sup);
        if (lane == 0) keep[i] = anysup ? 0 : 1;
        __syncthreads();
    }
    for (int t = lane; t < N; t += 32) keepO[order[t]] = keep[t];
}

// ---------------------------------------------------------------------------
// 5) Final output: recompute gathered sigmoid, multiply by keep, and append
//    keep flags (as 0/1 floats) after the mask block.
// ---------------------------------------------------------------------------
__global__ __launch_bounds__(256)
void finalize_kernel(const float* __restrict__ x,
                     const int* __restrict__ labels,
                     const int* __restrict__ keepO,
                     float* __restrict__ out,
                     int C, int K, int N, long long out_size) {
    int n = blockIdx.x;
    int c = labels[n];
    float kf = keepO[n] ? 1.0f : 0.0f;
    const float* src = x + ((size_t)n * C + c) * K;
    float* dst = out + (size_t)n * K;
    for (int k = threadIdx.x; k < K; k += 256)
        dst[k] = kf / (1.0f + __expf(-src[k]));
    long long kpos = (long long)N * K + n;
    if (threadIdx.x == 0 && kpos < out_size)
        out[kpos] = kf;
}

extern "C" void kernel_launch(void* const* d_in, const int* in_sizes, int n_in,
                              void* d_out, int out_size, void* d_ws, size_t ws_size,
                              hipStream_t stream) {
    const float* x      = (const float*)d_in[0];
    const int*   labels = (const int*)d_in[1];
    const float* scores = (const float*)d_in[2];

    const int N = in_sizes[1];                 // 512
    const int K = 56 * 56;                     // 3136 = 98 * 32
    const int C = in_sizes[0] / (N * K);       // 81

    // Workspace layout (~4.3 MB total)
    char*     ws    = (char*)d_ws;
    size_t    sB    = (size_t)N * K * sizeof(_Float16);
    _Float16* S     = (_Float16*)ws;                 // [N][K] f16, sorted order
    float*    area  = (float*)(ws + sB);             // [N] f32 (sorted order)
    float*    iou   = area + N;                      // [N][N] f32
    int*      order = (int*)(iou + (size_t)N * N);   // [N]
    int*      rank  = order + N;                     // [N]
    int*      keepO = rank + N;                      // [N] keep in original order

    rank_order_kernel<<<(N + 255) / 256, 256, 0, stream>>>(scores, rank, order, N);
    gather_sigmoid_kernel<<<N, 256, 0, stream>>>(x, labels, order, S, area, C, K);

    dim3 tiles(N / 16, N / 16);
    inter_iou_wmma_kernel<<<tiles, 32, 0, stream>>>(S, area, iou, N, K);

    greedy_nms_kernel<<<1, 32, 0, stream>>>(iou, order, keepO, N);

    finalize_kernel<<<N, 256, 0, stream>>>(x, labels, keepO, (float*)d_out,
                                           C, K, N, (long long)out_size);
}
